// AttnBlock3D_8804682957087
// MI455X (gfx1250) — matrix operand order repaired and checked by reading, not compiled, not run on hardware
//
#include <hip/hip_runtime.h>

typedef __attribute__((ext_vector_type(16))) _Float16 v16h;
typedef __attribute__((ext_vector_type(8)))  float    v8f;
typedef __attribute__((ext_vector_type(4)))  unsigned int v4u;

#define C_DIM 256
#define N_TOK 4096
#define B_SZ  2
#define NGRP  16

static __device__ inline int lane_id() { return (int)(threadIdx.x & 31u); }

// ---- WMMA fragment loaders (CDNA5 wave32 layouts, 16x16x32 f16) ----
// A-fragment: 16x32 tile (row m, K=32), row-major source, row stride ld halves.
// lanes 0-15: K = 0..7 (VGPR0-3) and 16..23 (VGPR4-7); lanes 16-31: K = 8..15 and 24..31.
static __device__ inline v16h load_frag_a(const _Float16* src, int ld) {
  const int l  = lane_id();
  const int m  = l & 15;
  const int kb = (l >> 4) << 3;               // 0 or 8
  const _Float16* p = src + (size_t)m * ld;
  v16h f;
  ((v4u*)&f)[0] = *(const v4u*)(p + kb);       // K = kb .. kb+7
  ((v4u*)&f)[1] = *(const v4u*)(p + 16 + kb);  // K = 16+kb .. 16+kb+7
  return f;
}

// B-fragment: 32x16 tile B[k][n]; source stored column-major tile[col][k]
// (K contiguous per column), column stride ld halves.
// lanes 0-15 hold K=0..7,16..23 of column (lane&15); lanes 16-31 hold K=8..15,24..31.
static __device__ inline v16h load_frag_b(const _Float16* src, int ld) {
  const int l  = lane_id();
  const int n  = l & 15;
  const int kb = (l >> 4) << 3;               // 0 or 8
  const _Float16* p = src + (size_t)n * ld + kb;
  v16h f;
  ((v4u*)&f)[0] = ((const v4u*)p)[0];
  ((v4u*)&f)[1] = ((const v4u*)(p + 16))[0];
  return f;
}

// D/C layout: element r of v8f <-> (M = r + 8*(lane>>4), N = lane&15)

static __device__ inline v8f wmma16(v16h a, v16h b, v8f c) {
  return __builtin_amdgcn_wmma_f32_16x16x32_f16(false, a, false, b, (short)0, c, false, false);
}

// acc += W(16x256 tile) * B(256x16), K-dim = 256 via 8 WMMAs
static __device__ inline v8f mm8(const _Float16* __restrict__ wbase, const v16h* bh) {
  v8f acc = {};
#pragma unroll
  for (int kc = 0; kc < 8; ++kc) {
    v16h a = load_frag_a(wbase + kc * 32, C_DIM);
    acc = wmma16(a, bh[kc], acc);
  }
  return acc;
}

// ---------------- 1) weights fp32 -> f16 ----------------
__global__ void cvt_w_kernel(const float* __restrict__ wq, const float* __restrict__ wk,
                             const float* __restrict__ wv, const float* __restrict__ wp,
                             _Float16* __restrict__ w16) {
  int idx = blockIdx.x * blockDim.x + threadIdx.x;
  if (idx < C_DIM * C_DIM) {
    w16[idx]                     = (_Float16)wq[idx];
    w16[C_DIM * C_DIM + idx]     = (_Float16)wk[idx];
    w16[2 * C_DIM * C_DIM + idx] = (_Float16)wv[idx];
    w16[3 * C_DIM * C_DIM + idx] = (_Float16)wp[idx];
  }
}

// ---------------- 2) GroupNorm -> ht [b][n][c] f16 (transposed) ----------------
__global__ __launch_bounds__(256) void gn_kernel(const float* __restrict__ x,
                                                 const float* __restrict__ gamma,
                                                 const float* __restrict__ beta,
                                                 _Float16* __restrict__ ht) {
  const int g = blockIdx.x, b = blockIdx.y;
  const int GSZ = (C_DIM / NGRP) * N_TOK;     // 16*4096 = 65536
  const float* xb = x + ((size_t)b * C_DIM + g * (C_DIM / NGRP)) * N_TOK;
  float s = 0.f, s2 = 0.f;
  for (int e = threadIdx.x; e < GSZ; e += blockDim.x) {
    float v = xb[e];
    s += v; s2 += v * v;
  }
  __shared__ float r1[256], r2[256];
  r1[threadIdx.x] = s; r2[threadIdx.x] = s2;
  __syncthreads();
  for (int off = 128; off > 0; off >>= 1) {
    if ((int)threadIdx.x < off) { r1[threadIdx.x] += r1[threadIdx.x + off];
                                  r2[threadIdx.x] += r2[threadIdx.x + off]; }
    __syncthreads();
  }
  const float mu = r1[0] * (1.0f / GSZ);
  const float var = r2[0] * (1.0f / GSZ) - mu * mu;
  const float rs = rsqrtf(var + 1e-6f);
  for (int e = threadIdx.x; e < GSZ; e += blockDim.x) {
    int ch = g * (C_DIM / NGRP) + (e >> 12);  // e / 4096
    int i  = e & (N_TOK - 1);
    float v = (xb[e] - mu) * rs * gamma[ch] + beta[ch];
    ht[((size_t)b * N_TOK + i) * C_DIM + ch] = (_Float16)v;
  }
}

// ---------------- 3) QKV projections (WMMA) ----------------
__global__ __launch_bounds__(32) void qkv_kernel(const _Float16* __restrict__ w16,
                                                 const _Float16* __restrict__ ht,
                                                 const float* __restrict__ bq,
                                                 const float* __restrict__ bk,
                                                 const float* __restrict__ bv,
                                                 _Float16* __restrict__ Qt,
                                                 _Float16* __restrict__ Kt,
                                                 _Float16* __restrict__ Vc) {
  const int ot = blockIdx.x, it = blockIdx.y, b = blockIdx.z;
  const int l = lane_id(), hi = l >> 4, nn = l & 15;
  const _Float16* htb = ht + ((size_t)b * N_TOK + it * 16) * C_DIM;
  v16h bh[8];
#pragma unroll
  for (int kc = 0; kc < 8; ++kc) bh[kc] = load_frag_b(htb + kc * 32, C_DIM);

  const int orow0 = ot * 16 + hi * 8;
  // Q (scaled by 1/sqrt(c) = 1/16), stored transposed [n][c]
  {
    v8f acc = mm8(w16 + (size_t)ot * 16 * C_DIM, bh);
    alignas(16) _Float16 tmp[8];
#pragma unroll
    for (int r = 0; r < 8; ++r) tmp[r] = (_Float16)((acc[r] + bq[orow0 + r]) * 0.0625f);
    *(v4u*)(Qt + ((size_t)b * N_TOK + it * 16 + nn) * C_DIM + orow0) = *(v4u*)tmp;
  }
  // K, stored transposed [n][c]
  {
    v8f acc = mm8(w16 + (size_t)C_DIM * C_DIM + (size_t)ot * 16 * C_DIM, bh);
    alignas(16) _Float16 tmp[8];
#pragma unroll
    for (int r = 0; r < 8; ++r) tmp[r] = (_Float16)(acc[r] + bk[orow0 + r]);
    *(v4u*)(Kt + ((size_t)b * N_TOK + it * 16 + nn) * C_DIM + orow0) = *(v4u*)tmp;
  }
  // V, stored [c][n]
  {
    v8f acc = mm8(w16 + (size_t)2 * C_DIM * C_DIM + (size_t)ot * 16 * C_DIM, bh);
#pragma unroll
    for (int r = 0; r < 8; ++r)
      Vc[((size_t)b * C_DIM + orow0 + r) * N_TOK + it * 16 + nn] = (_Float16)(acc[r] + bv[orow0 + r]);
  }
}

// ---------------- 4) Flash attention with async-to-LDS double buffering ----------------
// LDS plan (one block, byte offsets):
//   kS[2][32*256] f16 : 0      .. 32768   (32 keys x 256c, key-major)
//   vS[2][256*32] f16 : 32768  .. 65536   (256c x 32j, c-major)
//   sS[4][16*32] f32  : 65536  .. 73728   (per-wave S tile)
//   pS[4][16*32] f16  : 73728  .. 77824   (per-wave P tile)
//   stats 3*[4][16] f32: 77824 .. 78592
#define SM_KS 0
#define SM_VS 32768
#define SM_SS 65536
#define SM_PS 73728
#define SM_ST 77824
#define SM_TOTAL 78592

// Per-wave async staging of one 32-key K chunk + V chunk (16 x b128 per lane-set).
static __device__ inline void stage_kv_async(const _Float16* kt_src, const _Float16* vc_src,
                                             unsigned ldsK, unsigned ldsV, int wv, int l) {
#pragma unroll
  for (int i = 0; i < 8; ++i) {
    unsigned t = (unsigned)(wv * 256 + i * 32 + l);
    unsigned long long ga = (unsigned long long)(uintptr_t)kt_src + (unsigned long long)t * 16u;
    unsigned lo = ldsK + t * 16u;
    asm volatile("global_load_async_to_lds_b128 %0, %1, off"
                 :: "v"(lo), "v"(ga) : "memory");
  }
#pragma unroll
  for (int i = 0; i < 8; ++i) {
    unsigned t = (unsigned)(wv * 256 + i * 32 + l);
    unsigned c = t >> 2, part = t & 3;
    unsigned long long ga = (unsigned long long)(uintptr_t)vc_src +
                            (unsigned long long)c * (N_TOK * 2u) + part * 16u;
    unsigned lo = ldsV + t * 16u;
    asm volatile("global_load_async_to_lds_b128 %0, %1, off"
                 :: "v"(lo), "v"(ga) : "memory");
  }
}

__global__ __launch_bounds__(128) void attn_kernel(const _Float16* __restrict__ Qt,
                                                   const _Float16* __restrict__ Kt,
                                                   const _Float16* __restrict__ Vc,
                                                   _Float16* __restrict__ hot) {
  __shared__ __align__(16) unsigned char smem[SM_TOTAL];
  _Float16* kS = (_Float16*)(smem + SM_KS);
  _Float16* vS = (_Float16*)(smem + SM_VS);
  float*    sS = (float*)(smem + SM_SS);
  _Float16* pS = (_Float16*)(smem + SM_PS);
  float*    mrow = (float*)(smem + SM_ST);         // [4][16]
  float*    lrow = mrow + 64;                      // [4][16]
  float*    srow = lrow + 64;                      // [4][16]
  const unsigned smem_base = (unsigned)(uintptr_t)smem;  // LDS byte offset of block

  const int b  = blockIdx.y;
  const int wv = (int)(threadIdx.x >> 5);
  const int l = lane_id(), hi = l >> 4, nn = l & 15;
  const int i0 = blockIdx.x * 64 + wv * 16;

  // Q fragments live in registers for the whole pass (A: [16 i][256 c])
  v16h aq[8];
  const _Float16* qb = Qt + ((size_t)b * N_TOK + i0) * C_DIM;
#pragma unroll
  for (int kc = 0; kc < 8; ++kc) aq[kc] = load_frag_a(qb + kc * 32, C_DIM);

  v8f acc[16];
  const v8f zero = {};
#pragma unroll
  for (int ct = 0; ct < 16; ++ct) acc[ct] = zero;

  if (l < 16) { mrow[wv * 16 + l] = -1e30f; lrow[wv * 16 + l] = 0.f; }

  const _Float16* ktB = Kt + (size_t)b * N_TOK * C_DIM;
  const _Float16* vcB = Vc + (size_t)b * C_DIM * N_TOK;
  const unsigned ldsK0 = smem_base + SM_KS;
  const unsigned ldsV0 = smem_base + SM_VS;

  // prologue: async-stage chunk 0 into buffer 0
  stage_kv_async(ktB, vcB, ldsK0, ldsV0, wv, l);

  for (int j0 = 0; j0 < N_TOK; j0 += 32) {
    const int pb = (j0 >> 5) & 1;
    if (j0 + 32 < N_TOK) {
      // prefetch next chunk into other buffer while we compute on this one
      stage_kv_async(ktB + (size_t)(j0 + 32) * C_DIM, vcB + (j0 + 32),
                     ldsK0 + (unsigned)(pb ^ 1) * 16384u,
                     ldsV0 + (unsigned)(pb ^ 1) * 16384u, wv, l);
      // our own 16 older async ops (current buffer) are done when <=16 remain
      asm volatile("s_wait_asynccnt 16" ::: "memory");
    } else {
      asm volatile("s_wait_asynccnt 0" ::: "memory");
    }
    __syncthreads();   // all waves' current-buffer DMA visible

    _Float16* kCur = kS + (size_t)pb * (32 * C_DIM);
    _Float16* vCur = vS + (size_t)pb * (C_DIM * 32);

    // S = Q^T K for two 16-key sub-tiles (K-dim = 256 channels)
    float* sw = sS + wv * 512;
#pragma unroll
    for (int jj = 0; jj < 2; ++jj) {
      v8f s = zero;
#pragma unroll
      for (int kc = 0; kc < 8; ++kc) {
        v16h bk = load_frag_b(kCur + jj * 16 * C_DIM + kc * 32, C_DIM);
        s = wmma16(aq[kc], bk, s);
      }
#pragma unroll
      for (int r = 0; r < 8; ++r) sw[(hi * 8 + r) * 32 + jj * 16 + nn] = s[r];
    }

    // online softmax: lane t < 16 owns query row t
    if (l < 16) {
      const float* row = sw + l * 32;
      float mo = mrow[wv * 16 + l];
      float mx = mo;
      for (int j = 0; j < 32; ++j) mx = fmaxf(mx, row[j]);
      float sum = 0.f;
      _Float16* pw = pS + wv * 512 + l * 32;
      for (int j = 0; j < 32; ++j) {
        float e = __expf(row[j] - mx);
        sum += e;
        pw[j] = (_Float16)e;
      }
      float rsc = __expf(mo - mx);
      srow[wv * 16 + l] = rsc;
      lrow[wv * 16 + l] = lrow[wv * 16 + l] * rsc + sum;
      mrow[wv * 16 + l] = mx;
    }

    // O[c][i] accumulation: O = O*rescale(col i) + V * P^T
    v16h bp = load_frag_b(pS + wv * 512, 32);   // B: [32 j][16 i], P row-contig in j
    float fac = srow[wv * 16 + nn];             // rescale is per query column
#pragma unroll
    for (int ct = 0; ct < 16; ++ct) {
      v16h av = load_frag_a(vCur + ct * 16 * 32, 32);  // A: [16 c][32 j]
      v8f o = acc[ct];
#pragma unroll
      for (int r = 0; r < 8; ++r) o[r] *= fac;
      acc[ct] = wmma16(av, bp, o);
    }
    __syncthreads();   // everyone done reading current buffer before it is re-staged
  }

  // normalize by row sum (per query column) and store hot [n][c] f16
  float fin = 1.0f / lrow[wv * 16 + nn];
#pragma unroll
  for (int ct = 0; ct < 16; ++ct) {
    alignas(16) _Float16 tmp[8];
#pragma unroll
    for (int r = 0; r < 8; ++r) tmp[r] = (_Float16)(acc[ct][r] * fin);
    *(v4u*)(hot + ((size_t)b * N_TOK + i0 + nn) * C_DIM + ct * 16 + hi * 8) = *(v4u*)tmp;
  }
}

// ---------------- 5) output projection + bias + residual ----------------
__global__ __launch_bounds__(32) void proj_kernel(const _Float16* __restrict__ wp16,
                                                  const _Float16* __restrict__ hot,
                                                  const float* __restrict__ bpv,
                                                  const float* __restrict__ x,
                                                  float* __restrict__ out) {
  const int ot = blockIdx.x, it = blockIdx.y, b = blockIdx.z;
  const int l = lane_id(), hi = l >> 4, nn = l & 15;
  const _Float16* hb = hot + ((size_t)b * N_TOK + it * 16) * C_DIM;
  v16h bh[8];
#pragma unroll
  for (int kc = 0; kc < 8; ++kc) bh[kc] = load_frag_b(hb + kc * 32, C_DIM);
  v8f acc = mm8(wp16 + (size_t)ot * 16 * C_DIM, bh);
  const int orow0 = ot * 16 + hi * 8;
#pragma unroll
  for (int r = 0; r < 8; ++r) {
    size_t idx = ((size_t)b * C_DIM + orow0 + r) * N_TOK + it * 16 + nn;
    out[idx] = x[idx] + acc[r] + bpv[orow0 + r];
  }
}

extern "C" void kernel_launch(void* const* d_in, const int* in_sizes, int n_in,
                              void* d_out, int out_size, void* d_ws, size_t ws_size,
                              hipStream_t stream) {
  (void)in_sizes; (void)n_in; (void)out_size; (void)ws_size;
  const float* x     = (const float*)d_in[0];
  const float* gamma = (const float*)d_in[1];
  const float* beta  = (const float*)d_in[2];
  const float* wq    = (const float*)d_in[3];
  const float* bq    = (const float*)d_in[4];
  const float* wk    = (const float*)d_in[5];
  const float* bk    = (const float*)d_in[6];
  const float* wv    = (const float*)d_in[7];
  const float* bv    = (const float*)d_in[8];
  const float* wp    = (const float*)d_in[9];
  const float* bp    = (const float*)d_in[10];
  float* out = (float*)d_out;

  // workspace layout (f16): weights(4*64K) | ht | Qt | Kt | Vc | hot
  const size_t WELE = (size_t)C_DIM * C_DIM;        // 65536
  const size_t TELE = (size_t)B_SZ * N_TOK * C_DIM; // 2M halves per tensor
  _Float16* w16 = (_Float16*)d_ws;
  _Float16* ht  = w16 + 4 * WELE;
  _Float16* Qt  = ht + TELE;
  _Float16* Kt  = Qt + TELE;
  _Float16* Vc  = Kt + TELE;
  _Float16* hot = Vc + TELE;

  cvt_w_kernel<<<dim3(256), dim3(256), 0, stream>>>(wq, wk, wv, wp, w16);
  gn_kernel<<<dim3(NGRP, B_SZ), dim3(256), 0, stream>>>(x, gamma, beta, ht);
  qkv_kernel<<<dim3(C_DIM / 16, N_TOK / 16, B_SZ), dim3(32), 0, stream>>>(
      w16, ht, bq, bk, bv, Qt, Kt, Vc);
  attn_kernel<<<dim3(N_TOK / 64, B_SZ), dim3(128), 0, stream>>>(Qt, Kt, Vc, hot);
  proj_kernel<<<dim3(C_DIM / 16, N_TOK / 16, B_SZ), dim3(32), 0, stream>>>(
      w16 + 3 * WELE, hot, bp, x, out);
}
